// CausalSelfAttention_46858093199559
// MI455X (gfx1250) — matrix operand order repaired and checked by reading, not compiled, not run on hardware
//
#include <hip/hip_runtime.h>
#include <hip/hip_bf16.h>

typedef __attribute__((ext_vector_type(16))) __bf16 v16bf;
typedef __attribute__((ext_vector_type(8)))  __bf16 v8bf;
typedef __attribute__((ext_vector_type(8)))  float  v8f;
typedef __attribute__((ext_vector_type(4)))  unsigned int v4ui;
typedef __attribute__((ext_vector_type(8)))  int v8i;
typedef __attribute__((ext_vector_type(4)))  int v4i;

#define EMB   1024
#define HEADS 16
#define DH    64
#define SEQ   2048
#define BATCH 2
#define NTOK  (BATCH * SEQ)   // 4096

#if defined(__has_builtin)
#if __has_builtin(__builtin_amdgcn_tensor_load_to_lds) && __has_builtin(__builtin_amdgcn_s_wait_tensorcnt)
#define HAS_TDM 1
#endif
#endif
#ifndef HAS_TDM
#define HAS_TDM 0
#endif

// Toolchain arity probe for the TDM builtin (this toolchain: 6 args).
#if defined(__has_include)
#if __has_include(<hip/amd_detail/amd_gfx1250_TDM.h>)
#define TDM_SIX_ARGS 1
#endif
#endif
#ifndef TDM_SIX_ARGS
#define TDM_SIX_ARGS 1
#endif

// ---------------------------------------------------------------------------
// WMMA helpers (CDNA5 16x16x32 bf16, fp32 accumulate)
// ---------------------------------------------------------------------------
static __device__ __forceinline__ v8f wmma_bf16(v16bf a, v16bf b, v8f c) {
  return __builtin_amdgcn_wmma_f32_16x16x32_bf16(false, a, false, b, (short)0, c,
                                                 false, false);
}

// A fragment: 16(M) x 32(K), row-major source [M][ldk].
// Lanes 0-15 hold K 0..15, lanes 16-31 hold K 16..31 (contiguous, matches B).
static __device__ __forceinline__ v16bf load_a_frag(const __bf16* base, int ldk) {
  int lane = threadIdx.x & 31;
  int m    = lane & 15;
  int koff = (lane < 16) ? 0 : 16;
  const __bf16* p = base + m * ldk + koff;
  v8bf lo = *(const v8bf*)(p);
  v8bf hi = *(const v8bf*)(p + 8);
  v16bf r;
#pragma unroll
  for (int i = 0; i < 8; ++i) { r[i] = lo[i]; r[i + 8] = hi[i]; }
  return r;
}

// B fragment: 32(K) x 16(N), source is B^T row-major [N][ldk].
// Lanes 0-15: N=lane, K=0..15 ; lanes 16-31: N=lane-16, K=16..31 (contiguous).
static __device__ __forceinline__ v16bf load_b_frag(const __bf16* baseT, int ldk) {
  int lane = threadIdx.x & 31;
  int n    = lane & 15;
  int koff = (lane < 16) ? 0 : 16;
  return *(const v16bf*)(baseT + n * ldk + koff);
}

// ---------------------------------------------------------------------------
// Tensor Data Mover: 2D bf16 tile, global -> LDS (D# per ISA cdna5 ch.8).
// Issue from ONE wave only (EXEC ignored, wave-level DMA). Completion via
// s_wait_tensorcnt + workgroup barrier.
// ---------------------------------------------------------------------------
static __device__ __forceinline__ void tdm_load_2d(__bf16* ldst, const __bf16* gsrc,
                                                   unsigned rows, unsigned cols,
                                                   unsigned row_stride_elems) {
#if HAS_TDM
  unsigned lds_off = (unsigned)(unsigned long long)(size_t)ldst;  // low 32 bits = LDS offset
  unsigned long long ga = (unsigned long long)(size_t)gsrc;
  v4ui g0;
  g0[0] = 1u;                                            // count=1, user descriptor
  g0[1] = lds_off;                                       // lds_addr
  g0[2] = (unsigned)ga;                                  // global_addr[31:0]
  g0[3] = (unsigned)((ga >> 32) & 0x01ffffffu) | 0x80000000u;  // addr[56:32] | type=2
  v8i g1;
  g1[0] = 1 << 16;                                       // data_size=1 (2 bytes)
  g1[1] = (int)((cols & 0xffffu) << 16);                 // tensor_dim0[15:0]
  g1[2] = (int)((cols >> 16) | ((rows & 0xffffu) << 16));// tensor_dim0 hi | dim1 lo
  g1[3] = (int)((rows >> 16) | ((cols & 0xffffu) << 16));// dim1 hi | tile_dim0
  g1[4] = (int)(rows & 0xffffu);                         // tile_dim1 (tile_dim2=0)
  g1[5] = (int)row_stride_elems;                         // tensor_dim0_stride[31:0]
  g1[6] = 0;
  g1[7] = 0;
  v4i z4 = {0, 0, 0, 0};
#if TDM_SIX_ARGS
  v8i z8 = {0, 0, 0, 0, 0, 0, 0, 0};
  __builtin_amdgcn_tensor_load_to_lds(g0, g1, z4, z4, z8, 0);
#else
  __builtin_amdgcn_tensor_load_to_lds(g0, g1, z4, z4, 0);
#endif
#else
  // fallback: wave-cooperative copy (caller guards to one wave)
  int lane = threadIdx.x & 31;
  int cpr = cols >> 3;
  for (unsigned idx = lane; idx < rows * cpr; idx += 32) {
    unsigned r = idx / cpr, c = (idx % cpr) * 8;
    *(v8bf*)&ldst[r * cols + c] = *(const v8bf*)&gsrc[(size_t)r * row_stride_elems + c];
  }
#endif
}

static __device__ __forceinline__ void tdm_wait() {
#if HAS_TDM
  __builtin_amdgcn_s_wait_tensorcnt(0);
#endif
}

// ---------------------------------------------------------------------------
// Prep kernel A: X fp32 -> bf16 (one-time; removes cvt work from GEMM loops)
// ---------------------------------------------------------------------------
__global__ __launch_bounds__(256) void xcvt_kernel(const float* __restrict__ X,
                                                   __bf16* __restrict__ Y) {
  size_t i = ((size_t)blockIdx.x * 256 + threadIdx.x) * 8;
#pragma unroll
  for (int e = 0; e < 8; ++e) Y[i + e] = (__bf16)X[i + e];
}

// ---------------------------------------------------------------------------
// Prep kernel B: transpose weight block to bf16 [N][K] image (LDS-tiled).
// ---------------------------------------------------------------------------
__global__ __launch_bounds__(256) void wtrans_kernel(const float* __restrict__ W,
                                                     int ld, int col0,
                                                     __bf16* __restrict__ outT,
                                                     int outrow0) {
  __shared__ __bf16 tile[64][65];
  int kbase = blockIdx.x * 64, nbase = blockIdx.y * 64;
  int tx = threadIdx.x & 15, ty = threadIdx.x >> 4;
#pragma unroll
  for (int i = 0; i < 4; ++i)
#pragma unroll
    for (int j = 0; j < 4; ++j)
      tile[ty + 16 * i][tx + 16 * j] =
          (__bf16)W[(size_t)(kbase + ty + 16 * i) * ld + col0 + nbase + tx + 16 * j];
  __syncthreads();
#pragma unroll
  for (int i = 0; i < 4; ++i)
#pragma unroll
    for (int j = 0; j < 4; ++j)
      outT[(size_t)(outrow0 + nbase + ty + 16 * i) * EMB + kbase + tx + 16 * j] =
          tile[tx + 16 * j][ty + 16 * i];
}

// ---------------------------------------------------------------------------
// Shared GEMM K-step body: hoist ALL fragment loads (distinct regs, batched
// DS waits), then WMMA chain with >=4-op spacing between same-acc updates.
// ---------------------------------------------------------------------------
static __device__ __forceinline__ void gemm_step(const __bf16 (*As)[64],
                                                 const __bf16 (*Bts)[64],
                                                 int mblk, v8f acc[4]) {
  v16bf a0 = load_a_frag(&As[mblk * 16][0], 64);
  v16bf a1 = load_a_frag(&As[mblk * 16][32], 64);
  v16bf blo[4], bhi[4];
#pragma unroll
  for (int nb = 0; nb < 4; ++nb) {
    blo[nb] = load_b_frag(&Bts[nb * 16][0], 64);
    bhi[nb] = load_b_frag(&Bts[nb * 16][32], 64);
  }
#pragma unroll
  for (int nb = 0; nb < 4; ++nb) acc[nb] = wmma_bf16(a0, blo[nb], acc[nb]);
#pragma unroll
  for (int nb = 0; nb < 4; ++nb) acc[nb] = wmma_bf16(a1, bhi[nb], acc[nb]);
}

// ---------------------------------------------------------------------------
// Kernel 1: fused Q/V projection. K of QKV is dead code (att = q @ q^T), so
// only Q and V columns are computed. 128x64 tile, BK=64, 8 waves; wave =
// M-block, 8 WMMAs / wave / K-step. TDM double-buffered staging.
// Writes Q bf16 [B,H,T,Dh] and V^T bf16 [B,H,Dh,T].
// ---------------------------------------------------------------------------
__global__ __launch_bounds__(256) void qv_proj_kernel(
    const __bf16* __restrict__ Xbf, const __bf16* __restrict__ Wt,
    const float* __restrict__ bias, __bf16* __restrict__ qws,
    __bf16* __restrict__ vtws) {
  __shared__ __align__(16) __bf16 As[2][128][64];   // 32 KB
  __shared__ __align__(16) __bf16 Bts[2][64][64];   // 16 KB
  int tid = threadIdx.x, lane = tid & 31, wave = tid >> 5;
  int mbase   = blockIdx.x * 128;
  int nglbase = blockIdx.y * 64;
  int mblk = wave;
  v8f acc[4] = {};
  const __bf16* Abase = Xbf + (size_t)mbase * EMB;
  const __bf16* Bbase = Wt + (size_t)nglbase * EMB;

  if (wave == 0) {
    tdm_load_2d(&As[0][0][0], Abase, 128, 64, EMB);
    tdm_load_2d(&Bts[0][0][0], Bbase, 64, 64, EMB);
    tdm_wait();
  }
  __syncthreads();

  int buf = 0;
  for (int kk = 0; kk < EMB; kk += 64, buf ^= 1) {
    if (kk + 64 < EMB && wave == 0) {   // prefetch next tile into other buffer
      tdm_load_2d(&As[buf ^ 1][0][0], Abase + kk + 64, 128, 64, EMB);
      tdm_load_2d(&Bts[buf ^ 1][0][0], Bbase + kk + 64, 64, 64, EMB);
    }
    gemm_step(As[buf], Bts[buf], mblk, acc);
    if (wave == 0) tdm_wait();          // prefetch done before next-iter barrier
    __syncthreads();
  }

  int mloc = (lane >> 4) << 3;   // C layout: rows r (lanes 0-15) / r+8 (16-31)
  int ncol = lane & 15;
#pragma unroll
  for (int nb = 0; nb < 4; ++nb) {
    int ngl  = nglbase + nb * 16 + ncol;
    int wcol = (ngl < EMB) ? ngl : ngl + EMB;   // b_attn index (skip K block)
    float bv = bias[wcol];
#pragma unroll
    for (int r = 0; r < 8; ++r) {
      int row = mbase + mblk * 16 + r + mloc;
      int b = row >> 11, t = row & (SEQ - 1);
      float v = acc[nb][r] + bv;
      if (ngl < EMB) {
        int h = ngl >> 6, d = ngl & 63;
        qws[((size_t)(b * HEADS + h) * SEQ + t) * DH + d] = (__bf16)v;
      } else {
        int j = ngl - EMB; int h = j >> 6, d = j & 63;
        vtws[((size_t)(b * HEADS + h) * DH + d) * SEQ + t] = (__bf16)v;
      }
    }
  }
}

// ---------------------------------------------------------------------------
// Kernel 2: causal flash attention with S = Q Q^T. One WG = (b,h, 64 query
// rows); 4 waves; streams 32-key tiles through TDM-double-buffered LDS.
// ---------------------------------------------------------------------------
__global__ __launch_bounds__(128) void flash_attn_kernel(
    const __bf16* __restrict__ qws, const __bf16* __restrict__ vtws,
    __bf16* __restrict__ attout) {
  __shared__ __align__(16) __bf16 Qs[64][64];        // 8 KB
  __shared__ __align__(16) __bf16 Ks[2][32][64];     // 8 KB (key rows = Q rows)
  __shared__ __align__(16) __bf16 Vts[2][64][32];    // 8 KB (V^T: [d][key])
  __shared__ __align__(16) __bf16 Ps[4][16][32];     // 4 KB (C->A relayout)
  int tid = threadIdx.x, lane = tid & 31, wave = tid >> 5;
  int qt = blockIdx.x & 31;
  int bh = blockIdx.x >> 5;
  int qbase = qt * 64;
  const __bf16* qhead  = qws  + (size_t)bh * SEQ * DH;
  const __bf16* vthead = vtws + (size_t)bh * DH * SEQ;

  if (wave == 0) {
    tdm_load_2d(&Qs[0][0], qhead + (size_t)qbase * DH, 64, 64, DH);
    tdm_load_2d(&Ks[0][0][0], qhead, 32, 64, DH);
    tdm_load_2d(&Vts[0][0][0], vthead, 64, 32, SEQ);
    tdm_wait();
  }
  __syncthreads();

  int mblk = wave;
  v16bf aq0 = load_a_frag(&Qs[mblk * 16][0], 64);    // K = d 0..31
  v16bf aq1 = load_a_frag(&Qs[mblk * 16][32], 64);   // K = d 32..63
  v8f acc[4] = {};
  float m_r[8], l_r[8];
#pragma unroll
  for (int r = 0; r < 8; ++r) { m_r[r] = -3.0e38f; l_r[r] = 0.0f; }
  int mloc = (lane >> 4) << 3;
  int ncol = lane & 15;
  const float scale = 0.125f;    // 1/sqrt(64)
  int jmax = (qbase + 63) >> 5;

  for (int j = 0; j <= jmax; ++j) {
    int buf = j & 1;
    int kb = j * 32;
    if (j < jmax && wave == 0) {   // prefetch next key/value tile
      int kb2 = kb + 32;
      tdm_load_2d(&Ks[buf ^ 1][0][0], qhead + (size_t)kb2 * DH, 32, 64, DH);
      tdm_load_2d(&Vts[buf ^ 1][0][0], vthead + kb2, 64, 32, SEQ);
    }

    // hoist all S-tile B fragments (distinct regs -> batched DS waits)
    v16bf bk0  = load_b_frag(&Ks[buf][0][0], 64);
    v16bf bk1  = load_b_frag(&Ks[buf][16][0], 64);
    v16bf bk0h = load_b_frag(&Ks[buf][0][32], 64);
    v16bf bk1h = load_b_frag(&Ks[buf][16][32], 64);
    // hoist V fragments too: in flight during the softmax below
    v16bf bv[4];
#pragma unroll
    for (int nb = 0; nb < 4; ++nb) bv[nb] = load_b_frag(&Vts[buf][nb * 16][0], 32);

    // S tiles: 16(M) x 32(N keys); interleave so same-acc WMMAs are 2 apart
    v8f s0 = {}, s1 = {};
    s0 = wmma_bf16(aq0, bk0, s0);
    s1 = wmma_bf16(aq0, bk1, s1);
    s0 = wmma_bf16(aq1, bk0h, s0);
    s1 = wmma_bf16(aq1, bk1h, s1);

    // online softmax: row stats reduced across each 16-lane half (C layout)
#pragma unroll
    for (int r = 0; r < 8; ++r) {
      int row = qbase + mblk * 16 + r + mloc;
      float v0 = (kb + ncol      <= row) ? s0[r] * scale : -3.0e38f;
      float v1 = (kb + 16 + ncol <= row) ? s1[r] * scale : -3.0e38f;
      float mt = fmaxf(v0, v1);
#pragma unroll
      for (int off = 1; off < 16; off <<= 1) mt = fmaxf(mt, __shfl_xor(mt, off, 16));
      float mnew = fmaxf(m_r[r], mt);
      float p0 = __expf(v0 - mnew), p1 = __expf(v1 - mnew);
      float rs = p0 + p1;
#pragma unroll
      for (int off = 1; off < 16; off <<= 1) rs += __shfl_xor(rs, off, 16);
      float fac = __expf(m_r[r] - mnew);
      m_r[r] = mnew;
      l_r[r] = l_r[r] * fac + rs;
#pragma unroll
      for (int nb = 0; nb < 4; ++nb) acc[nb][r] *= fac;
      Ps[mblk][r + mloc][ncol]      = (__bf16)p0;   // relayout C -> A via LDS
      Ps[mblk][r + mloc][ncol + 16] = (__bf16)p1;
    }
    // O += P * V  (same-wave DS ops are in-order: no barrier needed for Ps)
    v16bf ap = load_a_frag(&Ps[mblk][0][0], 32);
#pragma unroll
    for (int nb = 0; nb < 4; ++nb) acc[nb] = wmma_bf16(ap, bv[nb], acc[nb]);

    if (wave == 0) tdm_wait();     // prefetch complete before buffer swap
    __syncthreads();
  }

  // normalize and store bf16 att-out [NTOK][EMB], channel = h*64 + d
  int b = bh >> 4, h = bh & 15;
#pragma unroll
  for (int r = 0; r < 8; ++r) {
    float inv = 1.0f / l_r[r];
    int row = b * SEQ + qbase + mblk * 16 + r + mloc;
#pragma unroll
    for (int nb = 0; nb < 4; ++nb) {
      int col = h * DH + nb * 16 + ncol;
      attout[(size_t)row * EMB + col] = (__bf16)(acc[nb][r] * inv);
    }
  }
}

// ---------------------------------------------------------------------------
// Kernel 3: output projection: attout(bf16) @ w_proj^T-image + b_proj -> fp32.
// ---------------------------------------------------------------------------
__global__ __launch_bounds__(256) void out_proj_kernel(
    const __bf16* __restrict__ A, const __bf16* __restrict__ Wt,
    const float* __restrict__ bias, float* __restrict__ out) {
  __shared__ __align__(16) __bf16 As[2][128][64];
  __shared__ __align__(16) __bf16 Bts[2][64][64];
  int tid = threadIdx.x, lane = tid & 31, wave = tid >> 5;
  int mbase = blockIdx.x * 128, nbase = blockIdx.y * 64;
  int mblk = wave;
  v8f acc[4] = {};
  const __bf16* Abase = A + (size_t)mbase * EMB;
  const __bf16* Bbase = Wt + (size_t)nbase * EMB;

  if (wave == 0) {
    tdm_load_2d(&As[0][0][0], Abase, 128, 64, EMB);
    tdm_load_2d(&Bts[0][0][0], Bbase, 64, 64, EMB);
    tdm_wait();
  }
  __syncthreads();

  int buf = 0;
  for (int kk = 0; kk < EMB; kk += 64, buf ^= 1) {
    if (kk + 64 < EMB && wave == 0) {
      tdm_load_2d(&As[buf ^ 1][0][0], Abase + kk + 64, 128, 64, EMB);
      tdm_load_2d(&Bts[buf ^ 1][0][0], Bbase + kk + 64, 64, 64, EMB);
    }
    gemm_step(As[buf], Bts[buf], mblk, acc);
    if (wave == 0) tdm_wait();
    __syncthreads();
  }

  int mloc = (lane >> 4) << 3, ncol = lane & 15;
#pragma unroll
  for (int nb = 0; nb < 4; ++nb) {
    int n = nbase + nb * 16 + ncol;
    float bv = bias[n];
#pragma unroll
    for (int r = 0; r < 8; ++r) {
      int row = mbase + mblk * 16 + r + mloc;
      out[(size_t)row * EMB + n] = acc[nb][r] + bv;
    }
  }
}

// ---------------------------------------------------------------------------
extern "C" void kernel_launch(void* const* d_in, const int* in_sizes, int n_in,
                              void* d_out, int out_size, void* d_ws, size_t ws_size,
                              hipStream_t stream) {
  const float* x      = (const float*)d_in[0];
  const float* w_attn = (const float*)d_in[1];
  const float* b_attn = (const float*)d_in[2];
  const float* w_proj = (const float*)d_in[3];
  const float* b_proj = (const float*)d_in[4];
  float* out = (float*)d_out;

  // workspace layout (all bf16):
  // xbf 8MB | wqvt 4MB | wpt 2MB | Q 8MB | V^T 8MB | att-out 8MB  = 38MB
  __bf16* xbf    = (__bf16*)d_ws;
  __bf16* wqvt   = xbf + (size_t)NTOK * EMB;
  __bf16* wpt    = wqvt + (size_t)2 * EMB * EMB;
  __bf16* qws    = wpt + (size_t)EMB * EMB;
  __bf16* vtws   = qws + (size_t)BATCH * HEADS * SEQ * DH;
  __bf16* attout = vtws + (size_t)BATCH * HEADS * SEQ * DH;

  // one-time precision/layout prep
  xcvt_kernel<<<(NTOK * EMB) / (256 * 8), 256, 0, stream>>>(x, xbf);
  wtrans_kernel<<<dim3(16, 16), 256, 0, stream>>>(w_attn, 3 * EMB, 0,       wqvt, 0);
  wtrans_kernel<<<dim3(16, 16), 256, 0, stream>>>(w_attn, 3 * EMB, 2 * EMB, wqvt, EMB);
  wtrans_kernel<<<dim3(16, 16), 256, 0, stream>>>(w_proj, EMB,     0,       wpt,  0);

  qv_proj_kernel<<<dim3(NTOK / 128, (2 * EMB) / 64), 256, 0, stream>>>(
      xbf, wqvt, b_attn, qws, vtws);
  flash_attn_kernel<<<dim3(BATCH * HEADS * (SEQ / 64)), 128, 0, stream>>>(
      qws, vtws, attout);
  out_proj_kernel<<<dim3(NTOK / 128, EMB / 64), 256, 0, stream>>>(
      attout, wpt, b_proj, out);
}
